// P2PNet_9955734192291
// MI455X (gfx1250) — compile-verified, matrix operand order checked by reading, not run on hardware
//
#include <hip/hip_runtime.h>
#include <hip/hip_bf16.h>

typedef __attribute__((ext_vector_type(16))) __bf16 v16bf;
typedef __attribute__((ext_vector_type(8)))  float  v8f;

#define BB    2
#define NPTS  4096
#define NQ    16384
#define CH1   64
#define CH2   128
#define CH3   256
#define AGGC  707
#define KPAD  736    // 23 * 32
#define L1KT  23
#define L1NT  16

static __device__ __forceinline__ unsigned short f2bf(float f) {
  unsigned int u = __float_as_uint(f);
  u += 0x7FFFu + ((u >> 16) & 1u);   // round-to-nearest-even
  return (unsigned short)(u >> 16);
}
static __device__ __forceinline__ float bf2f(unsigned short h) {
  return __uint_as_float(((unsigned int)h) << 16);
}

// ---- WMMA fragment helpers (wave32, 16x16x32 bf16) ----------------------
// A-matrix 16x32 fragment from a row-major bf16 tile in LDS.
// ISA layout: lane 0-15 -> M=lane, K pairs {half*8+0..7, 16+half*8+0..7}.
static __device__ __forceinline__ v16bf load_a(const unsigned short* Ms, int ld,
                                               int k0, int lane) {
  const int row  = lane & 15;
  const int half = lane >> 4;
  union { v16bf v; unsigned int u[8]; } fa;
  const unsigned int* p = (const unsigned int*)(Ms + row * ld);
  const int kb = (k0 + half * 8) >> 1;
#pragma unroll
  for (int j = 0; j < 4; ++j) fa.u[j] = p[kb + j];
#pragma unroll
  for (int j = 0; j < 4; ++j) fa.u[4 + j] = p[kb + 8 + j];
  return fa.v;
}

// B-matrix 32x16 fragment from the packed weight stream: 16 contiguous bf16
// per lane per fragment -> two global_load_b128 per fragment.
static __device__ __forceinline__ v16bf load_b(const unsigned short* P, int frag,
                                               int lane) {
  union { v16bf v; uint4 q[2]; } fb;
  const uint4* p = (const uint4*)(P + ((size_t)frag * 32 + lane) * 16);
  fb.q[0] = p[0];
  fb.q[1] = p[1];
  return fb.v;
}

static __device__ __forceinline__ v8f wmma_bf16(v16bf a, v16bf b, v8f c) {
  return __builtin_amdgcn_wmma_f32_16x16x32_bf16(false, a, false, b, (short)0, c,
                                                 false, false);
}

// ---- weight packing into fragment order --------------------------------
static __device__ __forceinline__ void pack_one(const float* src,
                                                unsigned short* dst, int e,
                                                int NT, int Kreal) {
  const int frag = e >> 9;            // 512 elems per fragment
  const int lane = (e >> 4) & 31;
  const int i    = e & 15;
  const int kt = frag / NT, nt = frag - kt * NT;
  const int half = lane >> 4;
  const int n = nt * 16 + (lane & 15);
  const int k = kt * 32 + ((i < 8) ? (half * 8 + i) : (16 + half * 8 + (i - 8)));
  const float v = (k < Kreal) ? src[(size_t)n * Kreal + k] : 0.0f;
  dst[e] = f2bf(v);
}

__global__ void pack_kernel(const float* r1, const float* r2, const float* r3,
                            const float* r4, const float* w2, const float* w3,
                            unsigned short* r1p, unsigned short* r2p,
                            unsigned short* r3p, unsigned short* r4b,
                            unsigned short* w2p, unsigned short* w3p,
                            int* gmax) {
  const int E1 = L1KT * L1NT * 512;   // 707(->736) x 256
  const int E2 = 8 * 8 * 512;         // 256 x 128
  const int E3 = 4 * 4 * 512;         // 128 x 64
  const int E4 = 2 * 8 * 512;         // 64  x 128 (FE)
  const int E5 = 4 * 16 * 512;        // 128 x 256 (FE)
  int e = blockIdx.x * 256 + threadIdx.x;
  if (e < E1) { pack_one(r1, r1p, e, L1NT, AGGC); return; }
  e -= E1;
  if (e < E2) { pack_one(r2, r2p, e, 8, 256); return; }
  e -= E2;
  if (e < E3) { pack_one(r3, r3p, e, 4, 128); return; }
  e -= E3;
  if (e < E4) { pack_one(w2, w2p, e, 8, 64); return; }
  e -= E4;
  if (e < E5) { pack_one(w3, w3p, e, 16, 128); return; }
  e -= E5;
  if (e < 64) { r4b[e] = f2bf(r4[e]); return; }
  e -= 64;
  if (e < BB * CH3) gmax[e] = 0;      // relu outputs >= 0 -> int-max punning ok
}

// ---- feature extractor: 3 -> 64 -> 128 -> 256 + global max -------------
__global__ void fe_kernel(const float* op, const float* w1, const float* b1,
                          const float* b2, const float* b3,
                          const unsigned short* w2p, const unsigned short* w3p,
                          unsigned short* f1b, unsigned short* f2b,
                          unsigned short* f3b, int* gmax) {
  __shared__ unsigned short f1t[16 * CH1];
  __shared__ unsigned short f2t[16 * CH2];
  const int tid  = threadIdx.x;
  const int lane = tid & 31;
  const int wave = tid >> 5;
  const int p0   = blockIdx.x * 16;   // 16 points per block, batch-aligned

  // stage 0: K=3 layer on VALU, write bf16 tile + feature table
  for (int idx = tid; idx < 16 * CH1; idx += 256) {
    const int r = idx >> 6, o = idx & 63;
    const int p  = p0 + r;
    const int bb = p / NPTS, pi = p - bb * NPTS;
    const float* base = op + (size_t)bb * 3 * NPTS;
    float x = base[pi], y = base[NPTS + pi], z = base[2 * NPTS + pi];
    float a = fmaf(w1[o * 3], x,
              fmaf(w1[o * 3 + 1], y, fmaf(w1[o * 3 + 2], z, b1[o])));
    a = a > 0.f ? a : 0.f;
    const unsigned short h = f2bf(a);
    f1t[idx] = h;
    f1b[(size_t)p * CH1 + o] = h;
  }
  __syncthreads();

  // stage 1: 64 -> 128 via WMMA (NT=8, one tile per wave)
  {
    const int nt = wave;
    v8f acc = {};
    for (int kt = 0; kt < 2; ++kt)
      acc = wmma_bf16(load_a(f1t, CH1, kt * 32, lane),
                      load_b(w2p, kt * 8 + nt, lane), acc);
    const int n = nt * 16 + (lane & 15);
    const int half = lane >> 4;
    const float bv = b2[n];
#pragma unroll
    for (int rr = 0; rr < 8; ++rr) {
      const int m = rr + 8 * half;
      float v = acc[rr] + bv;
      v = v > 0.f ? v : 0.f;
      const unsigned short h = f2bf(v);
      f2t[m * CH2 + n] = h;
      f2b[(size_t)(p0 + m) * CH2 + n] = h;
    }
  }
  __syncthreads();

  // stage 2: 128 -> 256 via WMMA (NT=16, two tiles per wave) + global max
  int* gm = gmax + (p0 / NPTS) * CH3;
#pragma unroll
  for (int t = 0; t < 2; ++t) {
    const int nt = wave + 8 * t;
    v8f acc = {};
    for (int kt = 0; kt < 4; ++kt)
      acc = wmma_bf16(load_a(f2t, CH2, kt * 32, lane),
                      load_b(w3p, kt * 16 + nt, lane), acc);
    const int n = nt * 16 + (lane & 15);
    const int half = lane >> 4;
    const float bv = b3[n];
#pragma unroll
    for (int rr = 0; rr < 8; ++rr) {
      const int m = rr + 8 * half;
      float v = acc[rr] + bv;
      v = v > 0.f ? v : 0.f;
      f3b[(size_t)(p0 + m) * CH3 + n] = f2bf(v);
      atomicMax(&gm[n], __float_as_int(v));
    }
  }
}

// ---- brute-force 3-NN with LDS-cached originals ------------------------
__global__ void knn_kernel(const float* op, const float* qp, int* kidx,
                           float* kw) {
  __shared__ float xs[NPTS], ys[NPTS], zs[NPTS];
  const int gid = blockIdx.x * 256 + threadIdx.x;
  const int b = gid / NQ, qi = gid - b * NQ;
  const float* base = op + (size_t)b * 3 * NPTS;
  for (int i = threadIdx.x; i < NPTS; i += 256) {
    xs[i] = base[i];
    ys[i] = base[NPTS + i];
    zs[i] = base[2 * NPTS + i];
  }
  __syncthreads();
  const float* qb = qp + (size_t)b * 3 * NQ;
  const float qx = qb[qi], qy = qb[NQ + qi], qz = qb[2 * NQ + qi];
  float d0 = 3.4e38f, d1 = 3.4e38f, d2 = 3.4e38f;
  int i0 = 0, i1 = 0, i2 = 0;
  for (int i = 0; i < NPTS; ++i) {   // lockstep i -> LDS broadcast reads
    const float dx = xs[i] - qx, dy = ys[i] - qy, dz = zs[i] - qz;
    const float d = fmaf(dx, dx, fmaf(dy, dy, dz * dz));
    if (d < d2) {
      if (d < d0)      { d2 = d1; i2 = i1; d1 = d0; i1 = i0; d0 = d; i0 = i; }
      else if (d < d1) { d2 = d1; i2 = i1; d1 = d;  i1 = i; }
      else             { d2 = d;  i2 = i; }
    }
  }
  const float r0 = 1.f / (sqrtf(d0) + 1e-8f);
  const float r1 = 1.f / (sqrtf(d1) + 1e-8f);
  const float r2 = 1.f / (sqrtf(d2) + 1e-8f);
  const float s = 1.f / (r0 + r1 + r2);
  kidx[gid * 3 + 0] = i0; kidx[gid * 3 + 1] = i1; kidx[gid * 3 + 2] = i2;
  kw[gid * 3 + 0] = r0 * s; kw[gid * 3 + 1] = r1 * s; kw[gid * 3 + 2] = r2 * s;
}

// ---- fused gather + 4-layer regressor MLP (WMMA) -----------------------
__global__ void reg_kernel(const float* qp, const int* gmax, const int* kidx,
                           const float* kw, const unsigned short* f1b,
                           const unsigned short* f2b, const unsigned short* f3b,
                           const unsigned short* r1p, const float* rb1,
                           const unsigned short* r2p, const float* rb2,
                           const unsigned short* r3p, const float* rb3,
                           const unsigned short* r4b, const float* rb4,
                           float* out) {
  __shared__ unsigned short aggS[16 * KPAD];
  __shared__ unsigned short h1S[16 * 256];
  __shared__ unsigned short h2S[16 * 128];
  __shared__ unsigned short h3S[16 * 64];
  const int tid  = threadIdx.x;
  const int lane = tid & 31;
  const int wave = tid >> 5;
  const int row0 = blockIdx.x * 16;
  const int b = row0 / NQ;

  // stage A: build the 16x736 bf16 activation tile in LDS
  for (int idx = tid; idx < 16 * KPAD; idx += 256) {
    const int r = idx / KPAD, c = idx - r * KPAD;
    const int row = row0 + r;
    const int qi = row - b * NQ;
    float v = 0.f;
    if (c < 3) {
      v = qp[(size_t)b * 3 * NQ + c * NQ + qi];
    } else if (c < 451) {
      const unsigned short* tab; int ch, chn;
      if (c < 67)       { tab = f1b; ch = c - 3;   chn = CH1; }
      else if (c < 195) { tab = f2b; ch = c - 67;  chn = CH2; }
      else              { tab = f3b; ch = c - 195; chn = CH3; }
      float acc = 0.f;
#pragma unroll
      for (int j = 0; j < 3; ++j) {
        const int   id = kidx[row * 3 + j];
        const float w  = kw[row * 3 + j];
        acc = fmaf(w, bf2f(tab[(size_t)(b * NPTS + id) * chn + ch]), acc);
      }
      v = acc;
    } else if (c < AGGC) {
      v = __int_as_float(gmax[b * CH3 + (c - 451)]);
    }
    aggS[idx] = f2bf(v);
  }
  __syncthreads();

  // layer 1: 736 -> 256 (NT=16, two tiles per wave, K-loop 23)
#pragma unroll
  for (int t = 0; t < 2; ++t) {
    const int nt = wave + 8 * t;
    v8f acc = {};
    for (int kt = 0; kt < L1KT; ++kt)
      acc = wmma_bf16(load_a(aggS, KPAD, kt * 32, lane),
                      load_b(r1p, kt * L1NT + nt, lane), acc);
    const int n = nt * 16 + (lane & 15);
    const int half = lane >> 4;
    const float bv = rb1[n];
#pragma unroll
    for (int rr = 0; rr < 8; ++rr) {
      const float v = acc[rr] + bv;
      h1S[(rr + 8 * half) * 256 + n] = f2bf(v > 0.f ? v : 0.f);
    }
  }
  __syncthreads();

  // layer 2: 256 -> 128 (NT=8, one tile per wave)
  {
    const int nt = wave;
    v8f acc = {};
    for (int kt = 0; kt < 8; ++kt)
      acc = wmma_bf16(load_a(h1S, 256, kt * 32, lane),
                      load_b(r2p, kt * 8 + nt, lane), acc);
    const int n = nt * 16 + (lane & 15);
    const int half = lane >> 4;
    const float bv = rb2[n];
#pragma unroll
    for (int rr = 0; rr < 8; ++rr) {
      const float v = acc[rr] + bv;
      h2S[(rr + 8 * half) * 128 + n] = f2bf(v > 0.f ? v : 0.f);
    }
  }
  __syncthreads();

  // layer 3: 128 -> 64 (NT=4, waves 0-3; wave-granular gating keeps EXEC all-1)
  if (wave < 4) {
    const int nt = wave;
    v8f acc = {};
    for (int kt = 0; kt < 4; ++kt)
      acc = wmma_bf16(load_a(h2S, 128, kt * 32, lane),
                      load_b(r3p, kt * 4 + nt, lane), acc);
    const int n = nt * 16 + (lane & 15);
    const int half = lane >> 4;
    const float bv = rb3[n];
#pragma unroll
    for (int rr = 0; rr < 8; ++rr) {
      const float v = acc[rr] + bv;
      h3S[(rr + 8 * half) * 64 + n] = f2bf(v > 0.f ? v : 0.f);
    }
  }
  __syncthreads();

  // layer 4: 64 -> 1 dot product, one query per lane
  if (tid < 16) {
    float acc = rb4[0];
#pragma unroll
    for (int k = 0; k < 64; ++k)
      acc = fmaf(bf2f(h3S[tid * 64 + k]), bf2f(r4b[k]), acc);
    out[row0 + tid] = acc;
  }
}

extern "C" void kernel_launch(void* const* d_in, const int* in_sizes, int n_in,
                              void* d_out, int out_size, void* d_ws,
                              size_t ws_size, hipStream_t stream) {
  (void)in_sizes; (void)n_in; (void)out_size; (void)ws_size;
  const float* op  = (const float*)d_in[0];
  const float* qp  = (const float*)d_in[1];
  const float* w1  = (const float*)d_in[2];
  const float* b1  = (const float*)d_in[3];
  const float* w2  = (const float*)d_in[4];
  const float* b2  = (const float*)d_in[5];
  const float* w3  = (const float*)d_in[6];
  const float* b3  = (const float*)d_in[7];
  const float* r1  = (const float*)d_in[8];
  const float* rb1 = (const float*)d_in[9];
  const float* r2  = (const float*)d_in[10];
  const float* rb2 = (const float*)d_in[11];
  const float* r3  = (const float*)d_in[12];
  const float* rb3 = (const float*)d_in[13];
  const float* r4  = (const float*)d_in[14];
  const float* rb4 = (const float*)d_in[15];
  float* out = (float*)d_out;

  char* ws = (char*)d_ws;
  size_t off = 0;
  auto take = [&](size_t bytes) {
    char* p = ws + off;
    off = (off + bytes + 255) & ~(size_t)255;
    return p;
  };
  unsigned short* r1p = (unsigned short*)take((size_t)L1KT * L1NT * 512 * 2);
  unsigned short* r2p = (unsigned short*)take((size_t)8 * 8 * 512 * 2);
  unsigned short* r3p = (unsigned short*)take((size_t)4 * 4 * 512 * 2);
  unsigned short* w2p = (unsigned short*)take((size_t)2 * 8 * 512 * 2);
  unsigned short* w3p = (unsigned short*)take((size_t)4 * 16 * 512 * 2);
  unsigned short* r4b = (unsigned short*)take(64 * 2);
  unsigned short* f1b = (unsigned short*)take((size_t)BB * NPTS * CH1 * 2);
  unsigned short* f2b = (unsigned short*)take((size_t)BB * NPTS * CH2 * 2);
  unsigned short* f3b = (unsigned short*)take((size_t)BB * NPTS * CH3 * 2);
  int*   gmax = (int*)take((size_t)BB * CH3 * 4);
  int*   kidx = (int*)take((size_t)BB * NQ * 3 * 4);
  float* kw   = (float*)take((size_t)BB * NQ * 3 * 4);

  const int packTotal = L1KT * L1NT * 512 + 8 * 8 * 512 + 4 * 4 * 512 +
                        2 * 8 * 512 + 4 * 16 * 512 + 64 + BB * CH3;
  pack_kernel<<<(packTotal + 255) / 256, 256, 0, stream>>>(
      r1, r2, r3, r4, w2, w3, r1p, r2p, r3p, r4b, w2p, w3p, gmax);
  fe_kernel<<<BB * NPTS / 16, 256, 0, stream>>>(op, w1, b1, b2, b3, w2p, w3p,
                                                f1b, f2b, f3b, gmax);
  knn_kernel<<<BB * NQ / 256, 256, 0, stream>>>(op, qp, kidx, kw);
  reg_kernel<<<BB * NQ / 16, 256, 0, stream>>>(qp, gmax, kidx, kw, f1b, f2b,
                                               f3b, r1p, rb1, r2p, rb2, r3p,
                                               rb3, r4b, rb4, out);
}
